// BahdanauAttention_13761075216826
// MI455X (gfx1250) — matrix operand order repaired and checked by reading, not compiled, not run on hardware
//
#include <hip/hip_runtime.h>
#include <hip/hip_bf16.h>
#include <math.h>

#define HID   512
#define BATCH 64
#define SEQ   4096
#define ROWS  (SEQ * BATCH)   // 262144 flattened (s,b) rows
#define KT    (HID / 32)      // 16 K-tiles
#define GT    (HID / 16)      // 32 g-tiles

typedef __attribute__((ext_vector_type(16))) __bf16          v16bf;
typedef __attribute__((ext_vector_type(16))) unsigned short  v16u;
typedef __attribute__((ext_vector_type(8)))  float           v8f;

#if __has_builtin(__builtin_amdgcn_tanhf)
#define fast_tanh(x) __builtin_amdgcn_tanhf(x)
#else
#define fast_tanh(x) tanhf(x)
#endif

__device__ __forceinline__ unsigned short f32_to_bf16(float f) {
    unsigned int u = __float_as_uint(f);
    unsigned int r = u + 0x7FFFu + ((u >> 16) & 1u);   // round-to-nearest-even
    return (unsigned short)(r >> 16);
}

__device__ __forceinline__ unsigned int pack_bf16x2(float lo, float hi) {
    unsigned int a = __float_as_uint(lo), b = __float_as_uint(hi);
    a = (a + 0x7FFFu + ((a >> 16) & 1u)) >> 16;
    b = (b + 0x7FFFu + ((b >> 16) & 1u)) & 0xFFFF0000u;
    return a | b;
}

// ---------------------------------------------------------------------------
// Kernel 0: prepack W2 (f32, row-major [g][h]) into bf16 WMMA B-fragments.
//   Fragment order: [kt][gt][lane][j]  (j = 0..15 bf16 per lane, 32B/lane)
//   lane: N = lane&15, half = lane>>4 ; element j -> k = kt*32 + half*16 + j
// ---------------------------------------------------------------------------
__global__ __launch_bounds__(256) void w2_prepack_kernel(
    const float* __restrict__ W2, unsigned short* __restrict__ w2bf)
{
    int base = (blockIdx.x * 256 + threadIdx.x) * 4;
    #pragma unroll
    for (int e = 0; e < 4; ++e) {
        int idx  = base + e;                    // 0 .. 262143
        int kt   = idx >> 14;                   // /(32*32*16)
        int gt   = (idx >> 9) & 31;
        int l    = (idx >> 4) & 31;
        int j    = idx & 15;
        int g    = gt * 16 + (l & 15);
        int k    = kt * 32 + ((l >> 4) << 4) + j;
        w2bf[idx] = f32_to_bf16(W2[(size_t)g * HID + k]);
    }
}

// ---------------------------------------------------------------------------
// Kernel 1: dec_proj[b,g] = sum_h decoder_hidden[b,h] * W1[g,h]   (64 x 512)
// ---------------------------------------------------------------------------
__global__ __launch_bounds__(256) void dec_proj_kernel(
    const float* __restrict__ dh, const float* __restrict__ W1,
    float* __restrict__ dp)
{
    const int b = blockIdx.x;
    const float4* x4 = (const float4*)(&dh[b * HID]);
    for (int g = threadIdx.x; g < HID; g += 256) {
        const float4* w4 = (const float4*)(&W1[(size_t)g * HID]);
        float s = 0.f;
        #pragma unroll 4
        for (int h = 0; h < HID / 4; ++h) {
            float4 a = x4[h], w = w4[h];
            s += a.x * w.x + a.y * w.y + a.z * w.z + a.w * w.w;
        }
        dp[b * HID + g] = s;
    }
}

// ---------------------------------------------------------------------------
// Kernel 2: fused  energy[b,s] = v . tanh(dec_proj[b,:] + enc[row,:] @ W2^T)
//   row = s*64 + b  (encoder_outputs flattened (S,B,H))
//   Block: 256 threads (8 waves), 64 rows x 512 g-cols, K=512 in 16 steps.
//   Wave w owns g-tiles [4w, 4w+3]; all waves share 4 row-tiles via LDS A.
//   A tile is double-buffered in LDS: fill(kt+1) overlaps the 16 WMMAs of kt,
//   with a single workgroup barrier per K-step.
// ---------------------------------------------------------------------------
__global__ __launch_bounds__(256) void energy_kernel(
    const float* __restrict__ enc,             // (ROWS, HID)
    const unsigned short* __restrict__ w2bf,   // prepacked B fragments
    const float* __restrict__ dec_proj,        // (BATCH, HID)
    const float* __restrict__ vvec,            // (HID)
    float* __restrict__ energy)                // (BATCH, SEQ)
{
    __shared__ __align__(32) unsigned int a_lds[2][4][32][8]; // double-buffered A frags
    __shared__ float e_red[64];

    const int tid      = threadIdx.x;
    const int lane     = tid & 31;
    const int wave     = tid >> 5;          // 0..7
    const int lhalf    = lane >> 4;         // 0/1
    const int lmod     = lane & 15;         // N within tile
    const int row_base = blockIdx.x * 64;

    if (tid < 64) e_red[tid] = 0.f;

    // per-thread constant part of the A-fill addressing (p-loop, 4 iters)
    auto fill_a = [&](int kt, int buf) {
        const int kb = kt * 32;
        #pragma unroll
        for (int p = tid; p < 1024; p += 256) {
            int rt = p >> 8;
            int f  = p & 255;
            int l  = f >> 3;                // fragment lane
            int q  = f & 7;                 // dword slot (elements 2q,2q+1)
            int j0 = 2 * q;
            int M  = l & 15;
            int K  = ((l >> 4) << 4) + j0; // contiguous k = 16*half + j (matches B prepack)
            const float2 f2 = *(const float2*)(
                &enc[(size_t)(row_base + rt * 16 + M) * HID + kb + K]);
            a_lds[buf][rt][l][q] = pack_bf16x2(f2.x, f2.y);
        }
    };

    v8f acc[4][4];                          // [row-tile][g-tile]
    #pragma unroll
    for (int rt = 0; rt < 4; ++rt)
        #pragma unroll
        for (int t = 0; t < 4; ++t)
            acc[rt][t] = (v8f){0.f,0.f,0.f,0.f,0.f,0.f,0.f,0.f};

    fill_a(0, 0);

    for (int kt = 0; kt < KT; ++kt) {
        const int cur = kt & 1;
        __syncthreads();                    // buf `cur` stores visible; reads of
                                            // buf `cur^1` from kt-1 drained
        if (kt + 1 < KT) fill_a(kt + 1, cur ^ 1);   // overlaps WMMAs below

        // --- B fragments: prepacked bf16, 32 contiguous bytes per lane
        v16bf bfrag[4];
        #pragma unroll
        for (int t = 0; t < 4; ++t) {
            size_t fidx = ((size_t)(kt * GT + (wave * 4 + t)) * 32 + lane) * 16;
            v16u bu = *(const v16u*)(&w2bf[fidx]);
            bfrag[t] = __builtin_bit_cast(v16bf, bu);
        }

        // --- 16 WMMAs: 4 row-tiles x 4 g-tiles, reusing A across g-tiles
        #pragma unroll
        for (int rt = 0; rt < 4; ++rt) {
            v16bf afrag = __builtin_bit_cast(v16bf,
                *(const v16u*)(&a_lds[cur][rt][lane][0]));
            #pragma unroll
            for (int t = 0; t < 4; ++t) {
                acc[rt][t] = __builtin_amdgcn_wmma_f32_16x16x32_bf16(
                    false, afrag, false, bfrag[t],
                    (short)0, acc[rt][t], false, false);
            }
        }
    }

    // --- epilogue: part[rt][i] = sum over this lane's g-columns of v*tanh(.)
    float part[4][8];
    #pragma unroll
    for (int rt = 0; rt < 4; ++rt)
        #pragma unroll
        for (int i = 0; i < 8; ++i) part[rt][i] = 0.f;

    #pragma unroll
    for (int t = 0; t < 4; ++t) {
        int g = (wave * 4 + t) * 16 + lmod;
        float vg = vvec[g];
        #pragma unroll
        for (int rt = 0; rt < 4; ++rt) {
            #pragma unroll
            for (int i = 0; i < 8; ++i) {
                int M   = rt * 16 + i + (lhalf << 3);   // C/D layout: VGPR i, half
                int row = row_base + M;
                int b   = row & (BATCH - 1);
                float e = fast_tanh(dec_proj[b * HID + g] + acc[rt][t][i]);
                part[rt][i] += vg * e;
            }
        }
    }
    // reduce across the 16 N-lanes of each half (M is constant within a half)
    #pragma unroll
    for (int off = 1; off < 16; off <<= 1)
        #pragma unroll
        for (int rt = 0; rt < 4; ++rt)
            #pragma unroll
            for (int i = 0; i < 8; ++i)
                part[rt][i] += __shfl_xor(part[rt][i], off, 32);

    if (lmod == 0) {
        #pragma unroll
        for (int rt = 0; rt < 4; ++rt)
            #pragma unroll
            for (int i = 0; i < 8; ++i)
                atomicAdd(&e_red[rt * 16 + i + (lhalf << 3)], part[rt][i]);
    }
    __syncthreads();

    if (tid < 64) {
        int row = row_base + tid;
        int b = row & (BATCH - 1);
        int s = row >> 6;
        energy[(size_t)b * SEQ + s] = e_red[tid];
    }
}

// ---------------------------------------------------------------------------
// Kernel 3: softmax over S per batch row; writes attn weights to d_out
// ---------------------------------------------------------------------------
__global__ __launch_bounds__(256) void softmax_kernel(
    const float* __restrict__ energy, float* __restrict__ attn)
{
    __shared__ float red[256];
    const int b = blockIdx.x, tid = threadIdx.x;
    const float* e = &energy[(size_t)b * SEQ];

    float m = -3.4e38f;
    for (int s = tid; s < SEQ; s += 256) m = fmaxf(m, e[s]);
    red[tid] = m; __syncthreads();
    for (int off = 128; off > 0; off >>= 1) {
        if (tid < off) red[tid] = fmaxf(red[tid], red[tid + off]);
        __syncthreads();
    }
    m = red[0]; __syncthreads();

    float sum = 0.f;
    for (int s = tid; s < SEQ; s += 256) sum += __expf(e[s] - m);
    red[tid] = sum; __syncthreads();
    for (int off = 128; off > 0; off >>= 1) {
        if (tid < off) red[tid] += red[tid + off];
        __syncthreads();
    }
    float inv = 1.f / red[0];

    for (int s = tid; s < SEQ; s += 256)
        attn[(size_t)b * SEQ + s] = __expf(e[s] - m) * inv;
}

// ---------------------------------------------------------------------------
// Kernel 4a: context partials over S chunks (bandwidth-bound 512 MB pass)
//   grid (BATCH, NCHUNK), 512 threads = h index
// ---------------------------------------------------------------------------
#define NCHUNK 8
#define CHUNK  (SEQ / NCHUNK)   // 512

__global__ __launch_bounds__(512) void context_partial_kernel(
    const float* __restrict__ enc, const float* __restrict__ attn,
    float* __restrict__ partial)  // (NCHUNK, BATCH, HID)
{
    const int b = blockIdx.x, c = blockIdx.y, h = threadIdx.x;
    float accv = 0.f;
    const int s0 = c * CHUNK;
    #pragma unroll 4
    for (int s = s0; s < s0 + CHUNK; ++s) {
        float w = attn[(size_t)b * SEQ + s];       // scalar (uniform) load
        accv += w * enc[((size_t)s * BATCH + b) * HID + h];
    }
    partial[((size_t)c * BATCH + b) * HID + h] = accv;
}

// Kernel 4b: reduce partials -> context in d_out[0 .. B*H)
__global__ __launch_bounds__(256) void context_reduce_kernel(
    const float* __restrict__ partial, float* __restrict__ ctx)
{
    int idx = blockIdx.x * 256 + threadIdx.x;      // over BATCH*HID = 32768
    float s = 0.f;
    #pragma unroll
    for (int c = 0; c < NCHUNK; ++c)
        s += partial[(size_t)c * BATCH * HID + idx];
    ctx[idx] = s;
}

// ---------------------------------------------------------------------------
extern "C" void kernel_launch(void* const* d_in, const int* in_sizes, int n_in,
                              void* d_out, int out_size, void* d_ws, size_t ws_size,
                              hipStream_t stream) {
    const float* dec_hidden = (const float*)d_in[0];   // (64, 512)
    const float* enc        = (const float*)d_in[1];   // (4096, 64, 512)
    const float* W1         = (const float*)d_in[2];   // (512, 512)
    const float* W2         = (const float*)d_in[3];   // (512, 512)
    const float* vvec       = (const float*)d_in[4];   // (512)

    float* out_ctx  = (float*)d_out;                         // (64, 512)
    float* out_attn = (float*)d_out + BATCH * HID;           // (64, 4096)

    // workspace layout
    char*  ws = (char*)d_ws;
    float*          dp      = (float*)(ws);                               // 128 KB
    float*          energy  = (float*)(ws + (size_t)BATCH * HID * 4);     // 1 MB
    float*          partial = (float*)(ws + (size_t)BATCH * HID * 4
                                          + (size_t)BATCH * SEQ * 4);     // 1 MB
    unsigned short* w2bf    = (unsigned short*)(ws + (size_t)BATCH * HID * 4
                                                   + (size_t)BATCH * SEQ * 4
                                                   + (size_t)NCHUNK * BATCH * HID * 4); // 512 KB

    w2_prepack_kernel<<<(HID * HID) / (256 * 4), 256, 0, stream>>>(W2, w2bf);

    dec_proj_kernel<<<BATCH, 256, 0, stream>>>(dec_hidden, W1, dp);

    energy_kernel<<<ROWS / 64, 256, 0, stream>>>(enc, w2bf, dp, vvec, energy);

    softmax_kernel<<<BATCH, 256, 0, stream>>>(energy, out_attn);

    context_partial_kernel<<<dim3(BATCH, NCHUNK), 512, 0, stream>>>(enc, out_attn, partial);

    context_reduce_kernel<<<(BATCH * HID) / 256, 256, 0, stream>>>(partial, out_ctx);
}
